// PatchEmbedding_38044820308624
// MI455X (gfx1250) — compile-verified, matrix operand order checked by reading, not run on hardware
//
#include <hip/hip_runtime.h>

typedef __attribute__((ext_vector_type(16))) __bf16        v16bf;
typedef __attribute__((ext_vector_type(8)))  float         v8f;
typedef __attribute__((ext_vector_type(4)))  unsigned int  u32x4;

#define IMG      224
#define CH       3
#define ROWPITCH (IMG * CH)       // 672 floats per image row
#define WIN      22
#define ROWB     (WIN * CH)       // 66 features per window row
#define KTOT     (WIN * WIN * CH) // 1452
#define KP       1472             // K padded to 23*64
#define NP       196              // patches per image (14*14)
#define MTOT     (64 * NP)        // 12544 output rows
#define ND       768              // output dim
#define BM       128
#define BN       128
#define BK       64
#define KS       72               // padded LDS k-stride (bf16 elems, 144B)
#define KITER    23               // KP / BK, exact

union BfPack8 { __bf16 h[8]; u32x4 q; };

// ---------------------------------------------------------------------------
// Pass 1a: im2col + f32->bf16.  A_bf[m][KP], zero-padded tail.
// ---------------------------------------------------------------------------
__global__ __launch_bounds__(256)
void im2col_bf16(const float* __restrict__ x, __bf16* __restrict__ A)
{
  const int t = blockIdx.x * 256 + threadIdx.x;
  if (t >= MTOT * (KP / 8)) return;
  const int kc = t % (KP / 8);
  const int m  = t / (KP / 8);

  const int b  = m / NP;
  const int p  = m - b * NP;
  const int py = p / 14;
  const int px = p - py * 14;
  int rs = py * 16 - 3; rs = rs < 0 ? 0 : (rs > IMG - WIN ? IMG - WIN : rs);
  int cs = px * 16 - 3; cs = cs < 0 ? 0 : (cs > IMG - WIN ? IMG - WIN : cs);
  const float* xb = x + (((long)b * IMG + rs) * IMG + cs) * CH;

  const int f0 = kc * 8;
  const int h  = f0 / ROWB;
  const int r  = f0 - h * ROWB;

  BfPack8 pk;
  #pragma unroll
  for (int j = 0; j < 8; ++j) {
    const int rr = r + j;                       // crosses row at most once (8 < 66)
    int off = h * ROWPITCH + rr + ((rr >= ROWB) ? (ROWPITCH - ROWB) : 0);
    const bool valid = (f0 + j) < KTOT;
    off = valid ? off : 0;
    const float v = xb[off];
    pk.h[j] = (__bf16)(valid ? v : 0.f);
  }
  *(u32x4*)&A[(long)m * KP + f0] = pk.q;
}

// ---------------------------------------------------------------------------
// Pass 1b: W[k][n] f32 -> Wt[n][k] bf16, zero-padded K tail.
// ---------------------------------------------------------------------------
__global__ __launch_bounds__(256)
void wtrans_bf16(const float* __restrict__ W, __bf16* __restrict__ Wt)
{
  const int t = blockIdx.x * 256 + threadIdx.x;
  if (t >= ND * (KP / 8)) return;
  const int n  = t % ND;          // adjacent threads -> coalesced W reads
  const int kc = t / ND;
  const int k0 = kc * 8;

  BfPack8 pk;
  #pragma unroll
  for (int j = 0; j < 8; ++j) {
    const int k = k0 + j;
    const bool v = k < KTOT;
    const float w = W[(long)(v ? k : 0) * ND + n];
    pk.h[j] = (__bf16)(v ? w : 0.f);
  }
  *(u32x4*)&Wt[(long)n * KP + k0] = pk.q;
}

// ---------------------------------------------------------------------------
// Pass 2: pure bf16 GEMM, 128x128 tile per block, BK=64, wave32 WMMA.
// ---------------------------------------------------------------------------
__global__ __launch_bounds__(256, 2)
void gemm_bf16_wmma(const __bf16* __restrict__ A, const __bf16* __restrict__ Wt,
                    float* __restrict__ out)
{
  __shared__ __bf16 As[BM * KS];   // [m][k]
  __shared__ __bf16 Bs[BN * KS];   // [n][k]

  const int tid  = threadIdx.x;
  const int lane = tid & 31;
  const int wave = tid >> 5;
  const int wm   = wave & 3;       // 4 waves along M
  const int wn   = wave >> 2;      // 2 waves along N

  const int n0 = blockIdx.x * BN;
  const int m0 = blockIdx.y * BM;

  // staging: each thread owns 32 contiguous k of one row (A) and one row (B)
  const int row  = tid >> 1;            // 0..127
  const int half = (tid & 1) * 32;      // k offset 0 or 32
  const __bf16* ag = A  + (long)(m0 + row) * KP + half;
  const __bf16* bg = Wt + (long)(n0 + row) * KP + half;
  __bf16* asd = &As[row * KS + half];
  __bf16* bsd = &Bs[row * KS + half];

  v8f acc[2][4];
  {
    v8f z = {0.f, 0.f, 0.f, 0.f, 0.f, 0.f, 0.f, 0.f};
    #pragma unroll
    for (int i = 0; i < 2; ++i)
      #pragma unroll
      for (int j = 0; j < 4; ++j)
        acc[i][j] = z;
  }

  const int kb = (lane >> 4) * 8;   // 0 or 8
  const int lr = lane & 15;

  for (int kt = 0; kt < KITER; ++kt) {
    // ---- stage tiles: 8 x global b128 -> 8 x ds_store_b128, no address math ----
    u32x4 a0 = *(const u32x4*)(ag);
    u32x4 a1 = *(const u32x4*)(ag + 8);
    u32x4 a2 = *(const u32x4*)(ag + 16);
    u32x4 a3 = *(const u32x4*)(ag + 24);
    u32x4 b0 = *(const u32x4*)(bg);
    u32x4 b1 = *(const u32x4*)(bg + 8);
    u32x4 b2 = *(const u32x4*)(bg + 16);
    u32x4 b3 = *(const u32x4*)(bg + 24);
    ((u32x4*)asd)[0] = a0;
    *(u32x4*)(asd + 8)  = a1;
    *(u32x4*)(asd + 16) = a2;
    *(u32x4*)(asd + 24) = a3;
    ((u32x4*)bsd)[0] = b0;
    *(u32x4*)(bsd + 8)  = b1;
    *(u32x4*)(bsd + 16) = b2;
    *(u32x4*)(bsd + 24) = b3;
    ag += BK; bg += BK;

    __syncthreads();

    // ---- compute: 2 K-chunks of 32, 8 WMMAs each ----
    #pragma unroll
    for (int c = 0; c < 2; ++c) {
      const int kc = c * 32 + kb;
      v16bf afrag[2];
      #pragma unroll
      for (int i = 0; i < 2; ++i) {
        const __bf16* src = &As[(wm * 32 + i * 16 + lr) * KS + kc];
        ((u32x4*)&afrag[i])[0] = *(const u32x4*)(src);        // K kc..kc+7
        ((u32x4*)&afrag[i])[1] = *(const u32x4*)(src + 16);   // K kc+16..kc+23
      }
      v16bf bfrag[4];
      #pragma unroll
      for (int j = 0; j < 4; ++j) {
        const __bf16* src = &Bs[(wn * 64 + j * 16 + lr) * KS + kc];
        ((u32x4*)&bfrag[j])[0] = *(const u32x4*)(src);
        ((u32x4*)&bfrag[j])[1] = *(const u32x4*)(src + 16);
      }
      #pragma unroll
      for (int i = 0; i < 2; ++i)
        #pragma unroll
        for (int j = 0; j < 4; ++j)
          acc[i][j] = __builtin_amdgcn_wmma_f32_16x16x32_bf16(
              false, afrag[i], false, bfrag[j],
              (short)0, acc[i][j], false, false);
    }

    __syncthreads();
  }

  // ---- epilogue: VGPR r -> M = r + 8*(lane/16), lane%16 -> N ----
  #pragma unroll
  for (int i = 0; i < 2; ++i) {
    const int gmr = m0 + wm * 32 + i * 16 + (lane >> 4) * 8;
    #pragma unroll
    for (int j = 0; j < 4; ++j) {
      const int gnc = n0 + wn * 64 + j * 16 + (lane & 15);
      float* o = out + (long)gmr * ND + gnc;
      #pragma unroll
      for (int r = 0; r < 8; ++r)
        o[(long)r * ND] = acc[i][j][r];
    }
  }
}

// ---------------------------------------------------------------------------
// Fallback: fused im2col+GEMM (used only if d_ws is too small).
// ---------------------------------------------------------------------------
__global__ __launch_bounds__(256, 2)
void patch_embed_fused(const float* __restrict__ x,
                       const float* __restrict__ W,
                       float* __restrict__ out)
{
  __shared__ __bf16 As[BM * KS];
  __shared__ __bf16 Bs[BN * KS];

  const int tid  = threadIdx.x;
  const int lane = tid & 31;
  const int wave = tid >> 5;
  const int wm   = wave & 3;
  const int wn   = wave >> 2;
  const int n0 = blockIdx.x * BN;
  const int m0 = blockIdx.y * BM;

  const int am    = tid >> 1;
  const int fbase = (tid & 1) * 32;
  const int gm    = m0 + am;
  const int b     = gm / NP;
  const int p     = gm - b * NP;
  const int py    = p / 14;
  const int px    = p - py * 14;
  int rs = py * 16 - 3; rs = rs < 0 ? 0 : (rs > IMG - WIN ? IMG - WIN : rs);
  int cs = px * 16 - 3; cs = cs < 0 ? 0 : (cs > IMG - WIN ? IMG - WIN : cs);
  const float* xbase = x + (((long)b * IMG + rs) * IMG + cs) * CH;

  int aoff = fbase, arem = fbase;
  const int bn = (tid & 31) * 4;
  const int bk = (tid >> 5) * 8;

  v8f acc[2][4];
  {
    v8f z = {0.f, 0.f, 0.f, 0.f, 0.f, 0.f, 0.f, 0.f};
    #pragma unroll
    for (int i = 0; i < 2; ++i)
      #pragma unroll
      for (int j = 0; j < 4; ++j)
        acc[i][j] = z;
  }

  for (int kt = 0; kt < KITER; ++kt) {
    const int k0 = kt * BK;
    const int nvalid = KTOT - (k0 + fbase);
    BfPack8 ap[4];
    #pragma unroll
    for (int j = 0; j < 32; ++j) {
      int off = aoff + j + (((arem + j) >= ROWB) ? (ROWPITCH - ROWB) : 0);
      const bool valid = (j < nvalid);
      off = valid ? off : 0;
      float v = xbase[off];
      ap[j >> 3].h[j & 7] = (__bf16)(valid ? v : 0.f);
    }
    {
      u32x4* dst = (u32x4*)&As[am * KS + fbase];
      dst[0] = ap[0].q; dst[1] = ap[1].q; dst[2] = ap[2].q; dst[3] = ap[3].q;
    }
    aoff += BK; arem += BK;
    if (arem >= ROWB) { arem -= ROWB; aoff += ROWPITCH - ROWB; }

    BfPack8 bp[4];
    #pragma unroll
    for (int kk = 0; kk < 8; ++kk) {
      const int k   = k0 + bk + kk;
      const bool kv = (k < KTOT);
      const long rw = (long)(kv ? k : 0) * ND;
      float4 w = *(const float4*)&W[rw + n0 + bn];
      if (!kv) { w.x = 0.f; w.y = 0.f; w.z = 0.f; w.w = 0.f; }
      bp[0].h[kk] = (__bf16)w.x;
      bp[1].h[kk] = (__bf16)w.y;
      bp[2].h[kk] = (__bf16)w.z;
      bp[3].h[kk] = (__bf16)w.w;
    }
    #pragma unroll
    for (int i = 0; i < 4; ++i)
      *(u32x4*)&Bs[(bn + i) * KS + bk] = bp[i].q;

    __syncthreads();

    const int kb = (lane >> 4) * 8;
    const int lr = lane & 15;
    #pragma unroll
    for (int c = 0; c < 2; ++c) {
      const int kc = c * 32 + kb;
      v16bf afrag[2];
      #pragma unroll
      for (int i = 0; i < 2; ++i) {
        const __bf16* src = &As[(wm * 32 + i * 16 + lr) * KS + kc];
        ((u32x4*)&afrag[i])[0] = *(const u32x4*)(src);
        ((u32x4*)&afrag[i])[1] = *(const u32x4*)(src + 16);
      }
      v16bf bfrag[4];
      #pragma unroll
      for (int j = 0; j < 4; ++j) {
        const __bf16* src = &Bs[(wn * 64 + j * 16 + lr) * KS + kc];
        ((u32x4*)&bfrag[j])[0] = *(const u32x4*)(src);
        ((u32x4*)&bfrag[j])[1] = *(const u32x4*)(src + 16);
      }
      #pragma unroll
      for (int i = 0; i < 2; ++i)
        #pragma unroll
        for (int j = 0; j < 4; ++j)
          acc[i][j] = __builtin_amdgcn_wmma_f32_16x16x32_bf16(
              false, afrag[i], false, bfrag[j],
              (short)0, acc[i][j], false, false);
    }

    __syncthreads();
  }

  #pragma unroll
  for (int i = 0; i < 2; ++i) {
    const int gmr = m0 + wm * 32 + i * 16 + (lane >> 4) * 8;
    #pragma unroll
    for (int j = 0; j < 4; ++j) {
      const int gnc = n0 + wn * 64 + j * 16 + (lane & 15);
      float* o = out + (long)gmr * ND + gnc;
      #pragma unroll
      for (int r = 0; r < 8; ++r)
        o[(long)r * ND] = acc[i][j][r];
    }
  }
}

// ---------------------------------------------------------------------------
extern "C" void kernel_launch(void* const* d_in, const int* in_sizes, int n_in,
                              void* d_out, int out_size, void* d_ws, size_t ws_size,
                              hipStream_t stream) {
  (void)in_sizes; (void)n_in; (void)out_size;
  const float* x = (const float*)d_in[0];   // [64,224,224,3] f32
  const float* W = (const float*)d_in[1];   // [1452,768] f32
  float* out = (float*)d_out;               // [64,196,768] f32

  const size_t bytesA  = (size_t)MTOT * KP * 2;   // 36.9 MB
  const size_t bytesWt = (size_t)ND * KP * 2;     //  2.3 MB
  dim3 gtile(ND / BN, MTOT / BM);                 // (6, 98)

  if (d_ws != nullptr && ws_size >= bytesA + bytesWt) {
    __bf16* Abf = (__bf16*)d_ws;
    __bf16* Wt  = (__bf16*)((char*)d_ws + bytesA);
    const int t1 = MTOT * (KP / 8);
    const int t2 = ND * (KP / 8);
    im2col_bf16<<<(t1 + 255) / 256, 256, 0, stream>>>(x, Abf);
    wtrans_bf16<<<(t2 + 255) / 256, 256, 0, stream>>>(W, Wt);
    gemm_bf16_wmma<<<gtile, dim3(256), 0, stream>>>(Abf, Wt, out);
  } else {
    patch_embed_fused<<<gtile, dim3(256), 0, stream>>>(x, W, out);
  }
}
